// FieldAttention_8469675507745
// MI455X (gfx1250) — compile-verified, matrix operand order checked by reading, not run on hardware
//
#include <hip/hip_runtime.h>
#include <cstdint>

// ---- dims (fixed by the reference) ----
#define B_      4
#define YX      64          // Y == X == 64
#define C_IN    512
#define C_CTX   512
#define C_MIX   128
#define NH      16          // heads
#define NG      4           // KV groups
#define HF      32
#define DD      64          // 2*HF, value dim
#define NQ      1024
#define NK      256
#define NV      256
#define NQKV    1536        // NQ+NK+NV
#define KIN     640         // C_IN + C_MIX
#define C_OUT   512
#define C_CAT   2048        // 2 * NH * DD
#define NTOK    16384       // B*Y*X

typedef __attribute__((ext_vector_type(16))) _Float16 v16h;
typedef __attribute__((ext_vector_type(8)))  float    v8f;

// ---------- WMMA helpers (CDNA5 layouts, ISA 7.12.2) ----------
// A-frag (16x32 f16): lane l -> row (l&15); K-halves: l<16 -> K {0..7,16..23},
// l>=16 -> K {8..15,24..31}. B-frag loaded identically from a B^T (N-major) tile.
__device__ __forceinline__ v16h load_frag(const _Float16* base, int r0, int stride, int kc) {
  int l = threadIdx.x & 31;
  const _Float16* p = base + (r0 + (l & 15)) * stride + kc + ((l & 16) ? 8 : 0);
  union { uint4 u[2]; v16h v; } x;
  x.u[0] = *(const uint4*)(p);
  x.u[1] = *(const uint4*)(p + 16);
  return x.v;
}

__device__ __forceinline__ v8f wmma16(v16h a, v16h b, v8f c) {
  // D = A*B + C, f32 accumulate
  return __builtin_amdgcn_wmma_f32_16x16x32_f16(false, a, false, b, (short)0, c, false, false);
}

// ---------- prep: mix = ctx@W_mix + b ; ckv = ctx@W_ctx + b ----------
__global__ void fa_prep(const float* __restrict__ context,
                        const float* __restrict__ W_mix, const float* __restrict__ b_mix,
                        const float* __restrict__ W_ctx, const float* __restrict__ b_ctx,
                        float* __restrict__ mix, float* __restrict__ ckv) {
  int tid = threadIdx.x;
  for (int o = tid; o < B_ * C_MIX; o += 256) {
    int b = o >> 7, n = o & 127;
    float s = b_mix[n];
    for (int k = 0; k < C_CTX; ++k) s += context[b * C_CTX + k] * W_mix[k * C_MIX + n];
    mix[o] = s;
  }
  for (int o = tid; o < B_ * (NK + NV); o += 256) {
    int b = o >> 9, n = o & 511;
    float s = b_ctx[n];
    for (int k = 0; k < C_CTX; ++k) s += context[b * C_CTX + k] * W_ctx[k * 512 + n];
    ckv[o] = s;
  }
}

// ---------- weight transpose + f32->f16 ----------
__global__ void fa_cvtw(const float* __restrict__ Wqkv, const float* __restrict__ Wout,
                        _Float16* __restrict__ Wqkv_t, _Float16* __restrict__ Wout_t) {
  int i = blockIdx.x * blockDim.x + threadIdx.x;
  int stride = gridDim.x * blockDim.x;
  for (int e = i; e < KIN * NQKV; e += stride) {
    int k = e / NQKV, n = e % NQKV;
    Wqkv_t[n * KIN + k] = (_Float16)Wqkv[e];      // [NQKV][KIN]  (B^T layout)
  }
  for (int e = i; e < C_CAT * C_OUT; e += stride) {
    int k = e / C_OUT, n = e % C_OUT;
    Wout_t[n * C_CAT + k] = (_Float16)Wout[e];    // [C_OUT][C_CAT]
  }
}

// ---------- QKV GEMM: [16384 x 640] x [640 x 1536] + bias -> qkv f16 ----------
// Double-buffered LDS; fixed per-thread load slots (no exec divergence).
__global__ void fa_qkv_gemm(const float* __restrict__ cells, const float* __restrict__ mix,
                            const _Float16* __restrict__ Wt, const float* __restrict__ bias,
                            _Float16* __restrict__ qkv) {
  __shared__ _Float16 As[2][128 * 32];   // [m][k]
  __shared__ _Float16 Bs[2][128 * 32];   // [n][k]  (B^T)
  const int m_block = blockIdx.x * 128;
  const int n_block = blockIdx.y * 128;
  const int tid = threadIdx.x, lane = tid & 31, wave = tid >> 5;
  const int wm = wave >> 2, wn = wave & 3;      // 2x4 wave grid: 64x32 per wave
  v8f acc[4][2] = {};
  float4 va[4];
  uint4  vb[2];

  auto loadA = [&](int kk) {
    // uniform source select: kk is a multiple of 32, boundary at C_MIX=128
#pragma unroll
    for (int u = 0; u < 4; ++u) {
      int e = tid + u * 256;                 // e < 1024 = 128*32/4
      int row = e >> 3, c4 = (e & 7) << 2;
      int tokr = m_block + row;
      va[u] = (kk < C_MIX)
          ? *(const float4*)(mix + (tokr >> 12) * C_MIX + kk + c4)
          : *(const float4*)(cells + (size_t)tokr * C_IN + (kk - C_MIX) + c4);
    }
  };
  auto loadB = [&](int kk) {
#pragma unroll
    for (int u = 0; u < 2; ++u) {
      int e = tid + u * 256;                 // e < 512 = 128*32/8
      int row = e >> 2, cv = (e & 3) << 3;
      vb[u] = *(const uint4*)(Wt + (size_t)(n_block + row) * KIN + kk + cv);
    }
  };
  auto stage = [&](int buf) {
#pragma unroll
    for (int u = 0; u < 4; ++u) {
      int e = tid + u * 256;
      _Float16* dst = &As[buf][e * 4];
      dst[0] = (_Float16)va[u].x; dst[1] = (_Float16)va[u].y;
      dst[2] = (_Float16)va[u].z; dst[3] = (_Float16)va[u].w;
    }
#pragma unroll
    for (int u = 0; u < 2; ++u)
      *(uint4*)(&Bs[buf][(tid + u * 256) * 8]) = vb[u];
  };

  loadA(0); loadB(0); stage(0);
  __syncthreads();
  const int NT = KIN / 32;                    // 20
  for (int t = 0; t < NT; ++t) {
    int cur = t & 1;
    if (t + 1 < NT) { loadA((t + 1) * 32); loadB((t + 1) * 32); }
    v16h bf[2];
#pragma unroll
    for (int j = 0; j < 2; ++j) bf[j] = load_frag(Bs[cur], wn * 32 + j * 16, 32, 0);
#pragma unroll
    for (int i = 0; i < 4; ++i) {
      v16h af = load_frag(As[cur], wm * 64 + i * 16, 32, 0);
#pragma unroll
      for (int j = 0; j < 2; ++j) acc[i][j] = wmma16(af, bf[j], acc[i][j]);
    }
    if (t + 1 < NT) stage(cur ^ 1);
    __syncthreads();
  }
  // epilogue: D layout -> lane n = l&15, rows r + 8*(l>=16)
#pragma unroll
  for (int i = 0; i < 4; ++i)
#pragma unroll
    for (int j = 0; j < 2; ++j) {
      int n = n_block + wn * 32 + j * 16 + (lane & 15);
      float bn = bias[n];
#pragma unroll
      for (int r = 0; r < 8; ++r) {
        int m = m_block + wm * 64 + i * 16 + r + ((lane >> 4) << 3);
        qkv[(size_t)m * NQKV + n] = (_Float16)(acc[i][j][r] + bn);
      }
    }
}

// ---------- axial attention: one block per (axis, b, line o, group); 4 heads inside ----------
__global__ void fa_attn(const _Float16* __restrict__ qkv, const float* __restrict__ ckv,
                        const float* __restrict__ xpos, const float* __restrict__ ypos,
                        const unsigned char* __restrict__ mask, const float* __restrict__ freqs,
                        _Float16* __restrict__ cat) {
  __shared__ _Float16 Kmat[80 * 64];   // [key][dim], key 0 = global ctx, 65..79 pad
  __shared__ _Float16 Vt[64 * 96];     // [dim][key], pads 0
  __shared__ _Float16 Qm[64 * 64];     // [q][dim], RoPE applied (per head)
  __shared__ float    Sc[64 * 80];     // scores f32 (per head)
  __shared__ _Float16 Pr[64 * 96];     // softmax probs, pads 0 (per head)
  const int bid = blockIdx.x;
  const int axis = bid >> 10;          // 0: attend along X (out_x), 1: along Y (out_y)
  const int b    = (bid >> 8) & 3;
  const int o    = (bid >> 2) & 63;
  const int kg   = bid & 3;
  const float* pos = (axis ? ypos : xpos) + b * YX * 2;
  const float* fr  = freqs + kg * HF * 2;
  const int tid = threadIdx.x, lane = tid & 31, wave = tid >> 5;
  auto tok = [&](int s) -> int { return axis ? ((b * YX + s) * YX + o) : ((b * YX + o) * YX + s); };

  // --- K (rotated, global key at row 0) : once per group ---
  for (int e = tid; e < 80 * 32; e += 256) {
    int j = e >> 5, f = e & 31;
    float r0 = 0.f, r1 = 0.f;
    if (j == 0) {
      r0 = ckv[b * 512 + kg * DD + f * 2 + 0];
      r1 = ckv[b * 512 + kg * DD + f * 2 + 1];
    } else if (j <= 64) {
      int s = j - 1;
      const _Float16* kp = qkv + (size_t)tok(s) * NQKV + NQ + kg * DD + f * 2;
      float k0 = (float)kp[0], k1 = (float)kp[1];
      float phi = pos[s * 2] * fr[f * 2] + pos[s * 2 + 1] * fr[f * 2 + 1];
      float sn, cs; __sincosf(phi, &sn, &cs);
      r0 = k0 * cs - k1 * sn; r1 = k0 * sn + k1 * cs;
    }
    Kmat[j * 64 + f * 2]     = (_Float16)r0;
    Kmat[j * 64 + f * 2 + 1] = (_Float16)r1;
  }
  // --- V^T (dim-major; global value at key 0) : once per group ---
  for (int e = tid; e < 64 * 96; e += 256) {
    int d = e / 96, j = e % 96;
    float v = 0.f;
    if (j == 0)        v = ckv[b * 512 + NK + kg * DD + d];
    else if (j <= 64)  v = (float)qkv[(size_t)tok(j - 1) * NQKV + NQ + NK + kg * DD + d];
    Vt[e] = (_Float16)v;
  }

  const int chbase = axis ? 0 : NH * DD;   // concat([out_y, out_x]) on channels

  for (int g = 0; g < 4; ++g) {
    const int head = kg * 4 + g;
    // --- Q (rotated) for this head ---
    for (int e = tid; e < 64 * 32; e += 256) {
      int qi = e >> 5, f = e & 31;
      const _Float16* qp = qkv + (size_t)tok(qi) * NQKV + head * DD + f * 2;
      float q0 = (float)qp[0], q1 = (float)qp[1];
      float phi = pos[qi * 2] * fr[f * 2] + pos[qi * 2 + 1] * fr[f * 2 + 1];
      float sn, cs; __sincosf(phi, &sn, &cs);
      Qm[qi * 64 + f * 2]     = (_Float16)(q0 * cs - q1 * sn);
      Qm[qi * 64 + f * 2 + 1] = (_Float16)(q0 * sn + q1 * cs);
    }
    __syncthreads();
    // --- scores: Q[64x64] x K^T -> Sc[64x80], scale + mask ---
    for (int job = wave; job < 20; job += 8) {
      int m0 = (job / 5) * 16, n0 = (job % 5) * 16;
      v8f acc = {};
#pragma unroll
      for (int kc = 0; kc < 64; kc += 32) {
        v16h a  = load_frag(Qm,   m0, 64, kc);
        v16h bb = load_frag(Kmat, n0, 64, kc);
        acc = wmma16(a, bb, acc);
      }
      int n = n0 + (lane & 15);
      bool vis;
      if (n == 0) vis = true;
      else if (n <= 64) {
        int s = n - 1;
        int midx = axis ? ((b * YX + s) * YX + o) : ((b * YX + o) * YX + s);
        vis = mask[midx] != 0;
      } else vis = false;
#pragma unroll
      for (int r = 0; r < 8; ++r) {
        int m = m0 + r + ((lane >> 4) << 3);
        Sc[m * 80 + n] = vis ? acc[r] * 0.125f : -3.0e38f;
      }
    }
    __syncthreads();
    // --- softmax over 65 real keys (pads are -3e38 -> exp 0) ---
    if (tid < 64) {
      float mx = -3.4e38f;
      for (int j = 0; j < 80; ++j) mx = fmaxf(mx, Sc[tid * 80 + j]);
      float sum = 0.f;
      for (int j = 0; j < 80; ++j) sum += __expf(Sc[tid * 80 + j] - mx);
      float inv = 1.f / sum;
      for (int j = 0; j < 80; ++j) Pr[tid * 96 + j] = (_Float16)(__expf(Sc[tid * 80 + j] - mx) * inv);
      for (int j = 80; j < 96; ++j) Pr[tid * 96 + j] = (_Float16)0.f;
    }
    __syncthreads();
    // --- out = P[64x96] x V[96x64] -> cat (f16) ---
    for (int job = wave; job < 16; job += 8) {
      int m0 = (job >> 2) * 16, n0 = (job & 3) * 16;
      v8f acc = {};
#pragma unroll
      for (int kc = 0; kc < 96; kc += 32) {
        v16h a  = load_frag(Pr, m0, 96, kc);
        v16h bb = load_frag(Vt, n0, 96, kc);
        acc = wmma16(a, bb, acc);
      }
      int n = n0 + (lane & 15);
#pragma unroll
      for (int r = 0; r < 8; ++r) {
        int m = m0 + r + ((lane >> 4) << 3);
        cat[(size_t)tok(m) * C_CAT + chbase + head * DD + n] = (_Float16)acc[r];
      }
    }
    __syncthreads();   // protect Qm/Sc/Pr before next head
  }
}

// ---------- output GEMM: cat[16384 x 2048](f16) x Wt_out -> d_out f32 ----------
__global__ void fa_out_gemm(const _Float16* __restrict__ cat, const _Float16* __restrict__ Wt,
                            const float* __restrict__ bias, float* __restrict__ out) {
  __shared__ _Float16 As[2][128 * 32];
  __shared__ _Float16 Bs[2][128 * 32];
  const int m_block = blockIdx.x * 128;
  const int n_block = blockIdx.y * 128;
  const int tid = threadIdx.x, lane = tid & 31, wave = tid >> 5;
  const int wm = wave >> 2, wn = wave & 3;
  v8f acc[4][2] = {};
  uint4 va[2], vb[2];

  auto loadT = [&](int kk) {
#pragma unroll
    for (int u = 0; u < 2; ++u) {
      int e = tid + u * 256;
      int row = e >> 2, cv = (e & 3) << 3;
      va[u] = *(const uint4*)(cat + (size_t)(m_block + row) * C_CAT + kk + cv);
      vb[u] = *(const uint4*)(Wt  + (size_t)(n_block + row) * C_CAT + kk + cv);
    }
  };
  auto stage = [&](int buf) {
#pragma unroll
    for (int u = 0; u < 2; ++u) {
      int e = (tid + u * 256) * 8;
      *(uint4*)(&As[buf][e]) = va[u];
      *(uint4*)(&Bs[buf][e]) = vb[u];
    }
  };

  loadT(0); stage(0);
  __syncthreads();
  const int NT = C_CAT / 32;                  // 64
  for (int t = 0; t < NT; ++t) {
    int cur = t & 1;
    if (t + 1 < NT) loadT((t + 1) * 32);
    v16h bf[2];
#pragma unroll
    for (int j = 0; j < 2; ++j) bf[j] = load_frag(Bs[cur], wn * 32 + j * 16, 32, 0);
#pragma unroll
    for (int i = 0; i < 4; ++i) {
      v16h af = load_frag(As[cur], wm * 64 + i * 16, 32, 0);
#pragma unroll
      for (int j = 0; j < 2; ++j) acc[i][j] = wmma16(af, bf[j], acc[i][j]);
    }
    if (t + 1 < NT) stage(cur ^ 1);
    __syncthreads();
  }
#pragma unroll
  for (int i = 0; i < 4; ++i)
#pragma unroll
    for (int j = 0; j < 2; ++j) {
      int n = n_block + wn * 32 + j * 16 + (lane & 15);
      float bn = bias[n];
#pragma unroll
      for (int r = 0; r < 8; ++r) {
        int m = m_block + wm * 64 + i * 16 + r + ((lane >> 4) << 3);
        out[(size_t)m * C_OUT + n] = acc[i][j][r] + bn;
      }
    }
}

extern "C" void kernel_launch(void* const* d_in, const int* in_sizes, int n_in,
                              void* d_out, int out_size, void* d_ws, size_t ws_size,
                              hipStream_t stream) {
  const float* cells   = (const float*)d_in[0];
  const float* context = (const float*)d_in[1];
  const float* xpos    = (const float*)d_in[2];
  const float* ypos    = (const float*)d_in[3];
  const unsigned char* mask = (const unsigned char*)d_in[4];
  const float* freqs   = (const float*)d_in[5];
  const float* W_mix   = (const float*)d_in[6];
  const float* b_mix   = (const float*)d_in[7];
  const float* W_qkv   = (const float*)d_in[8];
  const float* b_qkv   = (const float*)d_in[9];
  const float* W_ctx   = (const float*)d_in[10];
  const float* b_ctx   = (const float*)d_in[11];
  const float* W_out   = (const float*)d_in[12];
  const float* b_out   = (const float*)d_in[13];

  char* ws = (char*)d_ws;
  size_t off = 0;
  float*    mix    = (float*)(ws + off);      off += (size_t)B_ * C_MIX * 4;            // 2 KB
  float*    ckv    = (float*)(ws + off);      off += (size_t)B_ * 512 * 4;              // 8 KB
  _Float16* Wqkv_t = (_Float16*)(ws + off);   off += (size_t)NQKV * KIN * 2;            // 1.9 MB
  _Float16* Wout_t = (_Float16*)(ws + off);   off += (size_t)C_OUT * C_CAT * 2;         // 2 MB
  _Float16* qkv    = (_Float16*)(ws + off);   off += (size_t)NTOK * NQKV * 2;           // 50 MB
  _Float16* cat    = (_Float16*)(ws + off);   off += (size_t)NTOK * C_CAT * 2;          // 67 MB
  (void)ws_size; (void)n_in; (void)in_sizes; (void)out_size;

  fa_prep<<<1, 256, 0, stream>>>(context, W_mix, b_mix, W_ctx, b_ctx, mix, ckv);
  fa_cvtw<<<1024, 256, 0, stream>>>(W_qkv, W_out, Wqkv_t, Wout_t);
  fa_qkv_gemm<<<dim3(NTOK / 128, NQKV / 128), 256, 0, stream>>>(cells, mix, Wqkv_t, b_qkv, qkv);
  fa_attn<<<2 * B_ * YX * NG, 256, 0, stream>>>(qkv, ckv, xpos, ypos, mask, freqs, cat);
  fa_out_gemm<<<dim3(NTOK / 128, C_OUT / 128), 256, 0, stream>>>(cat, Wout_t, b_out, (float*)d_out);
}
